// Model_3324304687108
// MI455X (gfx1250) — matrix operand order repaired and checked by reading, not compiled, not run on hardware
//
#include <hip/hip_runtime.h>
#include <hip/hip_bf16.h>
#include <math.h>

typedef __attribute__((ext_vector_type(16))) _Float16 v16h;
typedef __attribute__((ext_vector_type(8)))  _Float16 v8h;
typedef __attribute__((ext_vector_type(2)))  _Float16 v2h;
typedef __attribute__((ext_vector_type(8)))  float    v8f;

#define BATCH    32
#define SEQ_LEN  512
#define ENC_IN   64
#define D_MODEL  128
#define D_INNER  256
#define D_STATE  16
#define DT_RANK  8
#define PRED_LEN 96
#define BN       (BATCH * ENC_IN)   /* 2048 sequences */
#define MAXM     (BN * 64)          /* 131072 rows at patch_len=8 */
#define TWO_PI_F 6.28318530717958647692f
#define TILE_K   32

// ============================================================================
// WMMA GEMM:  C[M,N] = Ah[M,K](f16) @ W[K,N](f32) (+bias[N]) (+addsrc[M,N])
// Optional outputs: C (f32) and/or Ch (f16 mirror for the next GEMM's A).
//
// Block = 256 threads = 8 waves arranged 4(M) x 2(N); block tile 64(M)x128(N).
// Each wave: 16x64 strip = 1 A fragment + 4 B fragments -> 4 WMMA per K-step.
//  - A: packed f16 activations; fragment = 2x global_load_b128 (runs of 8
//    consecutive K per the f16 A layout). No per-step conversions.
//  - B: f32 weights staged per K-step into LDS *transposed into fragment
//    order* (sbT[col][k], f16, row stride 40 halfs = 80B, 16B aligned), so a
//    fragment read is one 32B contiguous v16h = 2x ds_load_b128, zero packing.
//    All 4 fragments are preloaded into distinct registers before the WMMA
//    chain so the scheduler can use partial s_wait_dscnt instead of draining.
//  - Interior-N blocks: coalesced float4 loads of two K-rows per unit, packed
//    v2h ds_store_b32. Edge blocks: gather-then-store (batched clamped loads,
//    one wait, selects, stores).
// EXEC stays all-ones through WMMA: OOB rows/cols clamp addresses; only the
// stores are guarded.
// ============================================================================
__global__ __launch_bounds__(256) void gemm_wmma_kernel(
    const _Float16* __restrict__ Ah, const float* __restrict__ W,
    const float* __restrict__ bias, const float* __restrict__ addsrc,
    float* __restrict__ C, _Float16* __restrict__ Ch,
    int M, int N, int K) {
  __shared__ _Float16 sbT[128][40];      // [col][k] fragment-order, 10240 B

  const int tid  = threadIdx.x;
  const int lane = tid & 31;
  const int wave = tid >> 5;
  const int wm   = wave & 3;             // 0..3  (M sub-tile)
  const int wn   = wave >> 2;            // 0..1  (N sub-strip of 64)
  const int l16  = lane & 15;
  const int hi   = lane >> 4;            // 0 lanes 0-15, 1 lanes 16-31

  const int tile_m0 = blockIdx.x << 6;   // 64-row block tile
  const int tile_n0 = blockIdx.y << 7;   // 128-col block tile

  const int mrow = tile_m0 + (wm << 4) + l16;
  const int arow = (mrow < M) ? mrow : (M - 1);     // clamp; store is guarded
  const size_t abase = (size_t)arow * (size_t)K;    // in halfs
  const int ka_off = hi * 16;   // A in contiguous K order (k = 16*hi + e): K-half select
  const int kb_off = hi * 16;   // B in the same contiguous K order as A: K-half select

  const bool interiorN = (tile_n0 + 128 <= N) && ((N & 3) == 0);

  const v8f vzero = {0.f, 0.f, 0.f, 0.f, 0.f, 0.f, 0.f, 0.f};
  v8f accs[4];
#pragma unroll
  for (int f = 0; f < 4; ++f) accs[f] = vzero;

  int k0 = 0;
  // ---------------- full K tiles (no K predicates anywhere) ----------------
  for (; k0 + TILE_K <= K; k0 += TILE_K) {
    __syncthreads();                       // protect sbT from previous readers
    if (interiorN) {
      // 512 (2K-row x 4-col) units / 256 threads = 2 each; coalesced float4
      // loads, paired-K v2h stores (ds_store_b32).
#pragma unroll
      for (int j = 0; j < 2; ++j) {
        const int i   = tid + (j << 8);    // 0..511
        const int kk2 = (i >> 5) << 1;     // 0,2,..,30
        const int c4  = (i & 31) << 2;     // 0,4,..,124
        const float4 v0 =
            *(const float4*)(W + (size_t)(k0 + kk2) * (size_t)N +
                             (size_t)(tile_n0 + c4));
        const float4 v1 =
            *(const float4*)(W + (size_t)(k0 + kk2 + 1) * (size_t)N +
                             (size_t)(tile_n0 + c4));
        v2h p0; p0[0] = (_Float16)v0.x; p0[1] = (_Float16)v1.x;
        v2h p1; p1[0] = (_Float16)v0.y; p1[1] = (_Float16)v1.y;
        v2h p2; p2[0] = (_Float16)v0.z; p2[1] = (_Float16)v1.z;
        v2h p3; p3[0] = (_Float16)v0.w; p3[1] = (_Float16)v1.w;
        *(v2h*)&sbT[c4 + 0][kk2] = p0;
        *(v2h*)&sbT[c4 + 1][kk2] = p1;
        *(v2h*)&sbT[c4 + 2][kk2] = p2;
        *(v2h*)&sbT[c4 + 3][kk2] = p3;
      }
    } else {
      // edge blocks / small N: gather-then-store, batched loads, one wait
      float vals[16];
#pragma unroll
      for (int j = 0; j < 16; ++j) {
        const int idx = tid + (j << 8);    // 0..4095
        const int kk  = idx >> 7;
        const int cc  = idx & 127;
        const int gc  = tile_n0 + cc;
        const int gcc = (gc < N) ? gc : (N - 1);
        vals[j] = W[(size_t)(k0 + kk) * (size_t)N + (size_t)gcc];
      }
#pragma unroll
      for (int j = 0; j < 16; ++j) {
        const int idx = tid + (j << 8);
        const int kk  = idx >> 7;
        const int cc  = idx & 127;
        const int gc  = tile_n0 + cc;
        sbT[cc][kk] = (gc < N) ? (_Float16)vals[j] : (_Float16)0.f;
      }
    }
    __syncthreads();

    // A fragment: two runs of 8 packed halfs -> 2x global_load_b128
    const v8h a0 = *(const v8h*)(Ah + abase + (size_t)(k0 + ka_off));
    const v8h a1 = *(const v8h*)(Ah + abase + (size_t)(k0 + ka_off + 8));
    const v16h af = __builtin_shufflevector(a0, a1, 0, 1, 2, 3, 4, 5, 6, 7,
                                            8, 9, 10, 11, 12, 13, 14, 15);
    if (k0 + 2 * TILE_K <= K)   // speculative next-tile prefetch
      __builtin_prefetch(Ah + abase + (size_t)(k0 + TILE_K), 0, 1);

    // preload all 4 B fragments (8x ds_load_b128, independent destinations)
    v16h bfs[4];
#pragma unroll
    for (int f = 0; f < 4; ++f) {
      const int lc = (wn << 6) + (f << 4) + l16;   // local B column 0..127
      bfs[f] = *(const v16h*)&sbT[lc][kb_off];     // 2x ds_load_b128
    }
#pragma unroll
    for (int f = 0; f < 4; ++f)
      accs[f] = __builtin_amdgcn_wmma_f32_16x16x32_f16(
          false, af, false, bfs[f], (short)0, accs[f], false, false);
  }

  // ---------------- K tail (K%32 != 0: patch embed K=8/16) -----------------
  if (k0 < K) {
    __syncthreads();
    {
      float vals[16];
#pragma unroll
      for (int j = 0; j < 16; ++j) {
        const int idx = tid + (j << 8);
        const int kk  = idx >> 7;
        const int cc  = idx & 127;
        const int gk  = k0 + kk;
        const int gc  = tile_n0 + cc;
        const int gkc = (gk < K) ? gk : (K - 1);
        const int gcc = (gc < N) ? gc : (N - 1);
        vals[j] = W[(size_t)gkc * (size_t)N + (size_t)gcc];
      }
#pragma unroll
      for (int j = 0; j < 16; ++j) {
        const int idx = tid + (j << 8);
        const int kk  = idx >> 7;
        const int cc  = idx & 127;
        const int gk  = k0 + kk;
        const int gc  = tile_n0 + cc;
        sbT[cc][kk] =
            (gk < K && gc < N) ? (_Float16)vals[j] : (_Float16)0.f;
      }
    }
    __syncthreads();

    _Float16 avals[16];
#pragma unroll
    for (int e = 0; e < 16; ++e) {
      const int ka  = k0 + ka_off + e;
      const int kac = (ka < K) ? ka : (K - 1);
      avals[e] = Ah[abase + (size_t)kac];   // unconditional clamped load
    }
    v16h af;
#pragma unroll
    for (int e = 0; e < 16; ++e) {
      const int ka = k0 + ka_off + e;
      af[e] = (ka < K) ? avals[e] : (_Float16)0.f;   // select, no branch
    }
    v16h bfs[4];
#pragma unroll
    for (int f = 0; f < 4; ++f) {
      const int lc = (wn << 6) + (f << 4) + l16;
      bfs[f] = *(const v16h*)&sbT[lc][kb_off];
    }
#pragma unroll
    for (int f = 0; f < 4; ++f)
      accs[f] = __builtin_amdgcn_wmma_f32_16x16x32_f16(
          false, af, false, bfs[f], (short)0, accs[f], false, false);
  }

  // ---------------- epilogue: C/D layout N=lane%16, M=r+8*hi ---------------
#pragma unroll
  for (int f = 0; f < 4; ++f) {
    const int ncol = tile_n0 + (wn << 6) + (f << 4) + l16;
    if (ncol < N) {
      const float badd = bias ? bias[ncol] : 0.f;
#pragma unroll
      for (int r = 0; r < 8; ++r) {
        const int m = tile_m0 + (wm << 4) + r + (hi << 3);
        if (m < M) {
          float v = accs[f][r] + badd;
          if (addsrc) v += addsrc[(size_t)m * (size_t)N + (size_t)ncol];
          if (C)  C[(size_t)m * (size_t)N + (size_t)ncol] = v;
          if (Ch) Ch[(size_t)m * (size_t)N + (size_t)ncol] = (_Float16)v;
        }
      }
    }
  }
}

// ============================================================================
// RevIN: per (batch, channel) mean/std over L, normalize, transpose.
// x[B,L,N] -> xv[B*N,L] (f32, for router) + xvh (f16, GEMM-A mirror)
// ============================================================================
__global__ __launch_bounds__(256) void revin_kernel(
    const float* __restrict__ x, const float* __restrict__ rw,
    const float* __restrict__ rb, float* __restrict__ xv,
    _Float16* __restrict__ xvh, float* __restrict__ meanv,
    float* __restrict__ stdv) {
  const int bn = blockIdx.x;              // b*ENC_IN + n
  const int b = bn / ENC_IN, n = bn % ENC_IN;
  const int tid = threadIdx.x;
  __shared__ float ssum[256], ssq[256];
  float s = 0.f, sq = 0.f;
  for (int l = tid; l < SEQ_LEN; l += 256) {
    float v = x[((size_t)b * SEQ_LEN + l) * ENC_IN + n];
    s += v; sq += v * v;
  }
  ssum[tid] = s; ssq[tid] = sq; __syncthreads();
  for (int o = 128; o > 0; o >>= 1) {
    if (tid < o) { ssum[tid] += ssum[tid + o]; ssq[tid] += ssq[tid + o]; }
    __syncthreads();
  }
  const float m  = ssum[0] / (float)SEQ_LEN;
  const float va = ssq[0] / (float)SEQ_LEN - m * m;
  const float sd = sqrtf(va + 1e-5f);
  if (tid == 0) { meanv[bn] = m; stdv[bn] = sd; }
  const float w = rw[n], bb = rb[n];
  for (int l = tid; l < SEQ_LEN; l += 256) {
    float v = x[((size_t)b * SEQ_LEN + l) * ENC_IN + n];
    float o = (v - m) / sd * w + bb;
    xv[(size_t)bn * SEQ_LEN + l]  = o;
    xvh[(size_t)bn * SEQ_LEN + l] = (_Float16)o;
  }
}

// ============================================================================
// xe += pe_b[col] + pos[np,col] + node_embed[n,col]
// ============================================================================
__global__ __launch_bounds__(256) void add_embed_kernel(
    float* __restrict__ xe, const float* __restrict__ pe_b,
    const float* __restrict__ pos, const float* __restrict__ node,
    int npatch, size_t total) {
  size_t i = (size_t)blockIdx.x * 256 + threadIdx.x;
  if (i >= total) return;
  int col = (int)(i & 127);
  size_t r = i >> 7;
  int np = (int)(r % npatch);
  int n  = (int)((r / npatch) % ENC_IN);
  xe[i] += pe_b[col] + pos[(size_t)np * D_MODEL + col]
                     + node[(size_t)n * D_MODEL + col];
}

// ============================================================================
// Frequency router: per (bn, patch): rDFT energy -> 3 band sums -> normalize
// -> MLP(3 -> 32 -> 128) -> sigmoid. One 128-thread block per row.
// ============================================================================
__global__ __launch_bounds__(128) void router_kernel(
    const float* __restrict__ xv, const float* __restrict__ w1,
    const float* __restrict__ b1, const float* __restrict__ w2,
    const float* __restrict__ b2, float* __restrict__ gate,
    int patch_len, int npatch) {
  const int r  = blockIdx.x;            // bn*npatch + np
  const int np = r % npatch, bn = r / npatch;
  const float* xp = xv + (size_t)bn * SEQ_LEN + (size_t)np * patch_len;
  __shared__ float sx[32];
  __shared__ float senergy[17];
  __shared__ float se[3];
  __shared__ float sh[32];
  const int tid = threadIdx.x;
  if (tid < patch_len) sx[tid] = xp[tid];
  __syncthreads();
  const int fl = patch_len / 2 + 1;
  if (tid < fl) {
    float re = 0.f, im = 0.f;
    for (int t = 0; t < patch_len; ++t) {
      float ang = -TWO_PI_F * (float)(tid * t) / (float)patch_len;
      re += sx[t] * cosf(ang);
      im += sx[t] * sinf(ang);
    }
    senergy[tid] = re * re + im * im;
  }
  __syncthreads();
  if (tid < 3) {
    const int s1 = fl / 3, s2 = (2 * fl) / 3;
    const int lo  = (tid == 0) ? 0 : ((tid == 1) ? s1 : s2);
    const int hhi = (tid == 0) ? s1 : ((tid == 1) ? s2 : fl);
    float a = 0.f;
    for (int i = lo; i < hhi; ++i) a += senergy[i];
    se[tid] = a;
  }
  __syncthreads();
  const float denom = se[0] + se[1] + se[2] + 1e-6f;
  if (tid < 32) {
    float a = b1[tid];
    for (int i = 0; i < 3; ++i) a += (se[i] / denom) * w1[i * 32 + tid];
    sh[tid] = fmaxf(a, 0.f);
  }
  __syncthreads();
  float a = b2[tid];
  for (int j = 0; j < 32; ++j) a += sh[j] * w2[j * D_MODEL + tid];
  gate[(size_t)r * D_MODEL + tid] = 1.f / (1.f + expf(-a));
}

// ============================================================================
// xgh = (LayerNorm(xe) * gate) as f16 (GEMM-A mirror)
// ============================================================================
__global__ __launch_bounds__(128) void ln_gate_kernel(
    const float* __restrict__ xe, const float* __restrict__ g,
    const float* __restrict__ bta, const float* __restrict__ gate,
    _Float16* __restrict__ xgh) {
  const int r = blockIdx.x, tid = threadIdx.x;
  const size_t base = (size_t)r * D_MODEL;
  const float v = xe[base + tid];
  __shared__ float red[128];
  __shared__ float s_mean, s_var;
  red[tid] = v; __syncthreads();
  for (int o = 64; o > 0; o >>= 1) { if (tid < o) red[tid] += red[tid + o]; __syncthreads(); }
  if (tid == 0) s_mean = red[0] / 128.f;
  __syncthreads();
  const float c = v - s_mean;
  red[tid] = c * c; __syncthreads();
  for (int o = 64; o > 0; o >>= 1) { if (tid < o) red[tid] += red[tid + o]; __syncthreads(); }
  if (tid == 0) s_var = red[0] / 128.f;
  __syncthreads();
  const float out = c * rsqrtf(s_var + 1e-5f) * g[tid] + bta[tid];
  xgh[base + tid] = (_Float16)(out * gate[base + tid]);
}

// ============================================================================
// Depthwise causal conv (D_CONV=4) + SiLU.
// xz rows are [512] = [xi(256) | z(256)]; writes xc (f32, scan) + xch (f16).
// ============================================================================
__global__ __launch_bounds__(256) void conv_silu_kernel(
    const float* __restrict__ xz, const float* __restrict__ cw,
    const float* __restrict__ cb, float* __restrict__ xc,
    _Float16* __restrict__ xch, int npatch, size_t total) {
  size_t i = (size_t)blockIdx.x * 256 + threadIdx.x;
  if (i >= total) return;
  const int d = (int)(i & 255);
  const size_t r = i >> 8;
  const int t = (int)(r % npatch);
  const size_t bn = r / npatch;
  float acc = cb[d];
#pragma unroll
  for (int k = 0; k < 4; ++k) {
    int tt = t + k - 3;
    if (tt >= 0)
      acc += xz[((size_t)bn * npatch + tt) * 512 + d] * cw[d * 4 + k];
  }
  const float sv = acc / (1.f + expf(-acc));   // silu
  xc[i]  = sv;
  xch[i] = (_Float16)sv;
}

// ============================================================================
// dt = softplus(dbl[:, :8] @ dt_w + dt_b)   (K=8: plain VALU MACs)
// ============================================================================
__global__ __launch_bounds__(256) void dt_kernel(
    const float* __restrict__ dbl, const float* __restrict__ dtw,
    const float* __restrict__ dtb, float* __restrict__ dt, size_t total) {
  size_t i = (size_t)blockIdx.x * 256 + threadIdx.x;
  if (i >= total) return;
  const int d = (int)(i & 255);
  const size_t r = i >> 8;
  float acc = dtb[d];
#pragma unroll
  for (int j = 0; j < DT_RANK; ++j)
    acc += dbl[r * 40 + j] * dtw[j * D_INNER + d];
  dt[i] = (acc > 20.f) ? acc : log1pf(expf(acc));   // softplus
}

// ============================================================================
// Selective scan: one block (256 threads) per sequence; thread d owns channel
// d, 16-element hidden state in VGPRs; B_t/C_t broadcast via LDS.
// Fuses +u*Dp and *silu(z); writes f16 GEMM-A mirror.
// ============================================================================
__global__ __launch_bounds__(256) void scan_kernel(
    const float* __restrict__ dbl, const float* __restrict__ dtbuf,
    const float* __restrict__ xc, const float* __restrict__ xz,
    const float* __restrict__ Alog, const float* __restrict__ Dp,
    _Float16* __restrict__ ybh, int npatch) {
  const int bn = blockIdx.x;
  const int d  = threadIdx.x;
  float negA[D_STATE];
#pragma unroll
  for (int s = 0; s < D_STATE; ++s)
    negA[s] = -expf(Alog[(size_t)d * D_STATE + s]);   // A = -exp(Alog)
  float h[D_STATE];
#pragma unroll
  for (int s = 0; s < D_STATE; ++s) h[s] = 0.f;
  const float dpd = Dp[d];
  __shared__ float sB[D_STATE], sC[D_STATE];
  for (int t = 0; t < npatch; ++t) {
    const size_t row = (size_t)bn * npatch + t;
    __syncthreads();
    if (d < 16)      sB[d]      = dbl[row * 40 + DT_RANK + d];
    else if (d < 32) sC[d - 16] = dbl[row * 40 + DT_RANK + D_STATE + (d - 16)];
    __syncthreads();
    const float dtv = dtbuf[row * D_INNER + d];
    const float u   = xc[row * D_INNER + d];
    const float du  = dtv * u;
    float y = 0.f;
#pragma unroll
    for (int s = 0; s < D_STATE; ++s) {
      float dA = expf(dtv * negA[s]);
      h[s] = dA * h[s] + du * sB[s];
      y += h[s] * sC[s];
    }
    y += u * dpd;
    const float z = xz[row * 512 + D_INNER + d];
    y *= z / (1.f + expf(-z));                        // * silu(z)
    ybh[row * D_INNER + d] = (_Float16)y;
  }
}

// ============================================================================
// Final combine: softmax(scale_w)-weighted sum of 3 heads, inverse RevIN,
// transpose to out[b, t, n].
// ============================================================================
__global__ __launch_bounds__(256) void combine_kernel(
    const float* __restrict__ p0, const float* __restrict__ p1,
    const float* __restrict__ p2, const float* __restrict__ sw,
    const float* __restrict__ rw, const float* __restrict__ rb,
    const float* __restrict__ meanv, const float* __restrict__ stdv,
    float* __restrict__ out, size_t total) {
  size_t i = (size_t)blockIdx.x * 256 + threadIdx.x;
  if (i >= total) return;
  const int n = (int)(i % ENC_IN);
  const int t = (int)((i / ENC_IN) % PRED_LEN);
  const int b = (int)(i / ((size_t)ENC_IN * PRED_LEN));
  const float w0 = sw[0], w1 = sw[1], w2 = sw[2];
  const float mx = fmaxf(w0, fmaxf(w1, w2));
  const float e0 = expf(w0 - mx), e1 = expf(w1 - mx), e2 = expf(w2 - mx);
  const float es = e0 + e1 + e2;
  const size_t bn = (size_t)b * ENC_IN + n;
  float p = (p0[bn * PRED_LEN + t] * e0 + p1[bn * PRED_LEN + t] * e1 +
             p2[bn * PRED_LEN + t] * e2) / es;
  p = (p - rb[n]) / (rw[n] + 1e-10f);
  out[i] = p * stdv[bn] + meanv[bn];
}

// ============================================================================
// Host orchestration
// ============================================================================
static void launch_gemm(const _Float16* Ah, const float* W, const float* bias,
                        const float* addsrc, float* C, _Float16* Ch,
                        int M, int N, int K, hipStream_t s) {
  dim3 grid((unsigned)((M + 63) / 64), (unsigned)((N + 127) / 128));
  gemm_wmma_kernel<<<grid, 256, 0, s>>>(Ah, W, bias, addsrc, C, Ch, M, N, K);
}

extern "C" void kernel_launch(void* const* d_in, const int* in_sizes, int n_in,
                              void* d_out, int out_size, void* d_ws, size_t ws_size,
                              hipStream_t stream) {
  (void)in_sizes; (void)n_in; (void)out_size; (void)ws_size;
  const float* x       = (const float*)d_in[0];
  const float* revin_w = (const float*)d_in[1];
  const float* revin_b = (const float*)d_in[2];
  const float* node    = (const float*)d_in[3];
  const float* scale_w = (const float*)d_in[4];

  // workspace carve (256B-aligned blocks)
  char* wsb = (char*)d_ws;
  size_t off = 0;
  auto carve = [&](size_t bytes) -> char* {
    char* p = wsb + off;
    off += (bytes + 255) & ~(size_t)255;
    return p;
  };
  float*     xv   = (float*)    carve((size_t)BN * SEQ_LEN * 4);
  _Float16*  xvh  = (_Float16*) carve((size_t)BN * SEQ_LEN * 2);
  float*     mv   = (float*)    carve((size_t)BN * 4);
  float*     sv   = (float*)    carve((size_t)BN * 4);
  float*     xe   = (float*)    carve((size_t)MAXM * D_MODEL * 4);
  float*     gate = (float*)    carve((size_t)MAXM * D_MODEL * 4);
  _Float16*  xgh  = (_Float16*) carve((size_t)MAXM * D_MODEL * 2);
  float*     xz   = (float*)    carve((size_t)MAXM * 512 * 4);
  float*     xc   = (float*)    carve((size_t)MAXM * D_INNER * 4);
  _Float16*  xch  = (_Float16*) carve((size_t)MAXM * D_INNER * 2);
  float*     dbl  = (float*)    carve((size_t)MAXM * 40 * 4);
  float*     dtb  = (float*)    carve((size_t)MAXM * D_INNER * 4);
  _Float16*  ybh  = (_Float16*) carve((size_t)MAXM * D_INNER * 2);
  _Float16*  xoh  = (_Float16*) carve((size_t)MAXM * D_MODEL * 2);
  float*     pred = (float*)    carve((size_t)3 * BN * PRED_LEN * 4);

  // 1) RevIN + transpose (f32 + f16 mirror)
  revin_kernel<<<BN, 256, 0, stream>>>(x, revin_w, revin_b, xv, xvh, mv, sv);

  const int patch_lens[3] = {8, 16, 32};
  for (int s = 0; s < 3; ++s) {
    const int pl = patch_lens[s];
    const int npatch = SEQ_LEN / pl;
    const int M = BN * npatch;
    const float* const* P = (const float* const*)(d_in + 5 + 20 * s);
    const float *pe_w = P[0],  *pe_b = P[1],  *pos = P[2],  *r_w1 = P[3],
                *r_b1 = P[4],  *r_w2 = P[5],  *r_b2 = P[6], *ln_g = P[7],
                *ln_b = P[8],  *in_w = P[9],  *cw = P[10],  *cb = P[11],
                *xp_w = P[12], *dt_w = P[13], *dt_bias = P[14], *Alog = P[15],
                *Dpp = P[16],  *out_w = P[17],*head_w = P[18], *head_b = P[19];

    // 2) patch embedding GEMM: [M, pl] @ [pl, 128]   (bias added in step 3)
    launch_gemm(xvh, pe_w, nullptr, nullptr, xe, nullptr, M, D_MODEL, pl, stream);
    // 3) + pe_b + positional + node embeddings
    {
      size_t total = (size_t)M * D_MODEL;
      add_embed_kernel<<<(unsigned)((total + 255) / 256), 256, 0, stream>>>(
          xe, pe_b, pos, node, npatch, total);
    }
    // 4) router gate
    router_kernel<<<M, 128, 0, stream>>>(xv, r_w1, r_b1, r_w2, r_b2, gate, pl, npatch);
    // 5) layernorm * gate -> f16
    ln_gate_kernel<<<M, 128, 0, stream>>>(xe, ln_g, ln_b, gate, xgh);
    // 6) in_proj GEMM: [M,128] @ [128,512]
    launch_gemm(xgh, in_w, nullptr, nullptr, xz, nullptr, M, 2 * D_INNER,
                D_MODEL, stream);
    // 7) depthwise conv + silu (f32 + f16 mirror)
    {
      size_t total = (size_t)M * D_INNER;
      conv_silu_kernel<<<(unsigned)((total + 255) / 256), 256, 0, stream>>>(
          xz, cw, cb, xc, xch, npatch, total);
    }
    // 8) x_proj GEMM: [M,256] @ [256,40]
    launch_gemm(xch, xp_w, nullptr, nullptr, dbl, nullptr, M,
                DT_RANK + 2 * D_STATE, D_INNER, stream);
    // 9) dt = softplus(dbl[:, :8] @ dt_w + dt_b)
    {
      size_t total = (size_t)M * D_INNER;
      dt_kernel<<<(unsigned)((total + 255) / 256), 256, 0, stream>>>(
          dbl, dt_w, dt_bias, dtb, total);
    }
    // 10) selective scan (fuses +u*Dp and *silu(z)) -> f16
    scan_kernel<<<BN, 256, 0, stream>>>(dbl, dtb, xc, xz, Alog, Dpp, ybh, npatch);
    // 11) out_proj GEMM + residual: [M,256] @ [256,128] + xe -> f16 only
    launch_gemm(ybh, out_w, nullptr, xe, nullptr, xoh, M, D_MODEL, D_INNER,
                stream);
    // 12) head GEMM: [BN, npatch*128] @ [npatch*128, 96] + head_b
    launch_gemm(xoh, head_w, head_b, nullptr,
                pred + (size_t)s * BN * PRED_LEN, nullptr,
                BN, PRED_LEN, npatch * D_MODEL, stream);
  }

  // 13) combine scales + inverse RevIN + transpose to out[b, t, n]
  {
    size_t total = (size_t)BATCH * PRED_LEN * ENC_IN;
    combine_kernel<<<(unsigned)((total + 255) / 256), 256, 0, stream>>>(
        pred, pred + (size_t)BN * PRED_LEN, pred + (size_t)2 * BN * PRED_LEN,
        scale_w, revin_w, revin_b, mv, sv, (float*)d_out, total);
  }
}
